// MultiHeadAttention_32358283608319
// MI455X (gfx1250) — matrix operand order repaired and checked by reading, not compiled, not run on hardware
//
#include <hip/hip_runtime.h>
#include <hip/hip_bf16.h>
#include <math.h>
#include <cstdint>

// ---------------------------------------------------------------------------
// MI455X (gfx1250) GQA attention block.
// - All matmuls on v_wmma_f32_16x16x32_bf16 (f32 accumulate).
// - GEMM A-tiles fed by the Tensor Data Mover (tensor_load_to_lds + TENSORcnt).
// - Flash-attention K-tiles fed by global_load_async_to_lds_b128 (ASYNCcnt).
// - bf16 staging buffers in d_ws (~48 MB; fits in the 192 MB L2).
// ---------------------------------------------------------------------------

#define D_MODEL   2048
#define SEQ       2048
#define N_HEADS   32
#define N_KV      8
#define HEAD_DIM  64
#define KV_DIM    (N_KV * HEAD_DIM)   // 512

typedef __attribute__((ext_vector_type(16))) __bf16       v16bf;
typedef __attribute__((ext_vector_type(8)))  float        v8f;
typedef __attribute__((ext_vector_type(4)))  unsigned int u32x4;
typedef __attribute__((ext_vector_type(8)))  unsigned int u32x8;

// ---- WMMA fragment loaders (ISA 7.12.2 layouts, wave32) -------------------
// A 16x32 bf16: lane m = lane&15; half selects K ranges:
//   VGPR v<4 : K = 2v + 8*half ; VGPR v>=4: K = 16+2(v-4) + 8*half
__device__ __forceinline__ v16bf load_A_frag(const __bf16* base, int ld) {
  int lane = threadIdx.x & 31;
  int half = lane >> 4;
  int m    = lane & 15;
  const __bf16* row = base + (size_t)m * ld;
  v16bf a;
#pragma unroll
  for (int v = 0; v < 8; ++v) {
    int k = ((v < 4) ? (2 * v) : (16 + 2 * (v - 4))) + half * 8;
    a[2 * v]     = row[k];
    a[2 * v + 1] = row[k + 1];
  }
  return a;
}

// B 32x16 bf16, staged transposed [n][k]: lane n = lane&15; K same as A fragment.
__device__ __forceinline__ v16bf load_B_frag(const __bf16* baseT, int ld) {
  int lane = threadIdx.x & 31;
  int half = lane >> 4;
  int n    = lane & 15;
  const __bf16* col = baseT + (size_t)n * ld;
  v16bf b;
#pragma unroll
  for (int v = 0; v < 8; ++v) {
    int k = ((v < 4) ? (2 * v) : (16 + 2 * (v - 4))) + half * 8;
    b[2 * v]     = col[k];
    b[2 * v + 1] = col[k + 1];
  }
  return b;
}

// ---- fp32 -> bf16 conversion ----------------------------------------------
__global__ void cvt_kernel(const float* __restrict__ in, __bf16* __restrict__ out, int n) {
  int i = blockIdx.x * blockDim.x + threadIdx.x;
  if (i < n) out[i] = (__bf16)in[i];
}

// ---- RoPE (in-place on bf16 [S][H][64], interleaved pairs) ----------------
__global__ void rope_kernel(__bf16* __restrict__ x, const float* __restrict__ cf,
                            const float* __restrict__ sf, int H) {
  int idx = blockIdx.x * blockDim.x + threadIdx.x;
  int total = SEQ * H * (HEAD_DIM / 2);
  if (idx >= total) return;
  int j = idx & 31;
  int h = (idx >> 5) % H;
  int s = idx / (32 * H);
  __bf16* p = x + (((size_t)s * H + h) * HEAD_DIM + 2 * j);
  float xr = (float)p[0], xi = (float)p[1];
  float c = cf[s * 32 + j], sn = sf[s * 32 + j];
  p[0] = (__bf16)(xr * c - xi * sn);
  p[1] = (__bf16)(xr * sn + xi * c);
}

// ---- Tiled bf16 GEMM: C[M,N] = A[M,K] @ B[K,N], f32 accumulate ------------
// 128x128 block tile, 8 waves (4x2), wave tile 32x64, K-stage 64 (2 WMMA k-steps).
// A tile loaded by the Tensor Data Mover; B tile staged transposed by the waves.
#define MBLK 128
#define NBLK 128
#define KT   64
#define PAD  2

template <typename OutT>
__global__ __launch_bounds__(256) void gemm_bf16(
    const __bf16* __restrict__ A, const __bf16* __restrict__ B,
    OutT* __restrict__ C, int M, int N, int K) {
  __shared__ __bf16 As[MBLK][KT];           // TDM-packed rows, 128B stride
  __shared__ __bf16 Bs[NBLK][KT + PAD];     // transposed [n][k]
  int m0 = blockIdx.y * MBLK, n0 = blockIdx.x * NBLK;
  int tid = threadIdx.x, wave = tid >> 5, lane = tid & 31;
  int wr = wave >> 1, wc = wave & 1;        // 4x2 wave grid
  v8f acc[2][4] = {};

  const unsigned ldsA = (unsigned)(uintptr_t)&As[0][0];

  for (int k0 = 0; k0 < K; k0 += KT) {
    __syncthreads();

    // ---- A tile 128x64 via TDM (wave 0 issues; descriptors are uniform) ----
    if (wave == 0) {
      uint64_t gaddr = (uint64_t)(uintptr_t)A + ((uint64_t)m0 * (uint64_t)K + (uint64_t)k0) * 2u;
      u32x4 g0;
      g0[0] = 1u;                                        // count=1, 2D, no gather
      g0[1] = ldsA;                                      // LDS dest (bytes)
      g0[2] = (unsigned)gaddr;                           // global_addr[31:0]
      g0[3] = (unsigned)((gaddr >> 32) & 0x01FFFFFFu) | (2u << 30); // addr[56:32] | type=2
      u32x8 g1;
      g1[0] = (1u << 16);                                // data_size = 2 bytes
      g1[1] = ((unsigned)K & 0xFFFFu) << 16;             // tensor_dim0[15:0]
      g1[2] = (((unsigned)K >> 16) & 0xFFFFu) |          // tensor_dim0[31:16]
              (((unsigned)M & 0xFFFFu) << 16);           // tensor_dim1[15:0]
      g1[3] = (((unsigned)M >> 16) & 0xFFFFu) |          // tensor_dim1[31:16]
              ((unsigned)KT << 16);                      // tile_dim0 = 64
      g1[4] = (unsigned)MBLK;                            // tile_dim1 = 128, tile_dim2 = 0
      g1[5] = (unsigned)K;                               // tensor_dim0_stride[31:0]
      g1[6] = 0u;                                        // stride0 hi / stride1 lo
      g1[7] = 0u;
      asm volatile("tensor_load_to_lds %0, %1" :: "s"(g0), "s"(g1) : "memory");
      __builtin_amdgcn_s_wait_tensorcnt(0);
    }

    // ---- B tile 64x128, staged transposed [n][k] ----
    {
      int bc = tid & 15, kr = tid >> 4;
#pragma unroll
      for (int p = 0; p < 4; ++p) {
        int kk = kr + p * 16;
        const __bf16* src = B + (size_t)(k0 + kk) * N + n0 + bc * 8;
        uint4 u = *(const uint4*)src;
        const __bf16* hs = (const __bf16*)&u;
#pragma unroll
        for (int i = 0; i < 8; ++i) Bs[bc * 8 + i][kk] = hs[i];
      }
    }
    if (k0 + KT < K) {  // pull next B tile toward GL2 (global_prefetch_b8)
      __builtin_prefetch(B + (size_t)(k0 + KT + (tid >> 4)) * N + n0 + (tid & 15) * 8, 0, 0);
    }
    __syncthreads();

#pragma unroll
    for (int ks = 0; ks < 2; ++ks) {
      v16bf af0 = load_A_frag(&As[wr * 32][ks * 32],      KT);
      v16bf af1 = load_A_frag(&As[wr * 32 + 16][ks * 32], KT);
#pragma unroll
      for (int nt = 0; nt < 4; ++nt) {
        v16bf bf = load_B_frag(&Bs[wc * 64 + nt * 16][ks * 32], KT + PAD);
        acc[0][nt] = __builtin_amdgcn_wmma_f32_16x16x32_bf16(
            false, af0, false, bf, (short)0, acc[0][nt], false, false);
        acc[1][nt] = __builtin_amdgcn_wmma_f32_16x16x32_bf16(
            false, af1, false, bf, (short)0, acc[1][nt], false, false);
      }
    }
  }

  // C/D layout: m = r + 8*(lane>>4), n = lane&15
  int half = lane >> 4, n = lane & 15;
#pragma unroll
  for (int mt = 0; mt < 2; ++mt)
#pragma unroll
    for (int nt = 0; nt < 4; ++nt)
#pragma unroll
      for (int r = 0; r < 8; ++r) {
        int gm = m0 + wr * 32 + mt * 16 + half * 8 + r;
        int gn = n0 + wc * 64 + nt * 16 + n;
        C[(size_t)gm * N + gn] = (OutT)acc[mt][nt][r];
      }
}

// ---- Flash attention (causal, GQA), online softmax ------------------------
// grid (S/128, N_HEADS), 256 threads. Wave = 16 query rows. Key tiles of 32.
__global__ __launch_bounds__(256) void flash_attn(
    const __bf16* __restrict__ Q, const __bf16* __restrict__ Kv,
    const __bf16* __restrict__ Vv, __bf16* __restrict__ Out) {
  __shared__ __bf16 Ks[32][HEAD_DIM];             // [key][d], 128B rows (async B128)
  __shared__ __bf16 Vt[HEAD_DIM][32 + PAD];       // [d][key] for PV B-fragments
  __shared__ __bf16 Ps[8][16][32 + PAD];          // per-wave P, A-layout staging

  int h = blockIdx.y;
  int kvh = h >> 2;                               // N_REP = 4
  int qbase = blockIdx.x * 128;
  int tid = threadIdx.x, wave = tid >> 5, lane = tid & 31;
  int half = lane >> 4, nn = lane & 15;
  const float scale = 0.125f;                     // 1/sqrt(64)

  v16bf qf[2];
  {
    const __bf16* q0 = Q + ((size_t)(qbase + wave * 16) * N_HEADS + h) * HEAD_DIM;
    qf[0] = load_A_frag(q0,      N_HEADS * HEAD_DIM);
    qf[1] = load_A_frag(q0 + 32, N_HEADS * HEAD_DIM);
  }

  v8f o[4] = {};
  float mrun[8], lrun[8];
#pragma unroll
  for (int r = 0; r < 8; ++r) { mrun[r] = -INFINITY; lrun[r] = 0.f; }

  int nkb = (qbase + 128) / 32;                   // causal: keys <= block end
  for (int kb = 0; kb < nkb; ++kb) {
    // ---- K tile: async global->LDS (per-lane 16B chunks, ASYNCcnt) ----
    // ---- V tile: manual transposed staging ----
    {
      int key = tid >> 3, ch = (tid & 7) * 8;
      unsigned koff = (unsigned)((((unsigned)(kb * 32 + key) * N_KV + kvh) * HEAD_DIM + ch) * 2u);
      unsigned ldsK = (unsigned)(uintptr_t)&Ks[key][ch];
      asm volatile("global_load_async_to_lds_b128 %0, %1, %2"
                   :: "v"(ldsK), "v"(koff), "s"(Kv) : "memory");
      const __bf16* vsrc = Vv + ((size_t)(kb * 32 + key) * N_KV + kvh) * HEAD_DIM + ch;
      uint4 u = *(const uint4*)vsrc;
      const __bf16* hs = (const __bf16*)&u;
#pragma unroll
      for (int i = 0; i < 8; ++i) Vt[ch + i][key] = hs[i];
    }
    asm volatile("s_wait_asynccnt 0" ::: "memory");
    __syncthreads();

    // scores: 16 q x 32 keys, contraction over head_dim (2 k-steps)
    v8f sc[2] = {};
#pragma unroll
    for (int ks = 0; ks < 2; ++ks)
#pragma unroll
      for (int nt = 0; nt < 2; ++nt) {
        v16bf bf = load_B_frag(&Ks[nt * 16][ks * 32], HEAD_DIM);
        sc[nt] = __builtin_amdgcn_wmma_f32_16x16x32_bf16(
            false, qf[ks], false, bf, (short)0, sc[nt], false, false);
      }

    // scale + causal mask + row max (reduce across the 16-lane half-group)
    float rm[8];
#pragma unroll
    for (int r = 0; r < 8; ++r) {
      int qg = qbase + wave * 16 + half * 8 + r;
      float best = -INFINITY;
#pragma unroll
      for (int nt = 0; nt < 2; ++nt) {
        int kg = kb * 32 + nt * 16 + nn;
        float v = sc[nt][r] * scale;
        if (kg > qg) v = -INFINITY;
        sc[nt][r] = v;
        best = fmaxf(best, v);
      }
      rm[r] = best;
    }
#pragma unroll
    for (int msk = 1; msk <= 8; msk <<= 1)
#pragma unroll
      for (int r = 0; r < 8; ++r)
        rm[r] = fmaxf(rm[r], __shfl_xor(rm[r], msk, 32));

    // online softmax update; stage P (C-layout -> A-layout via LDS)
    float alpha[8], rs[8];
#pragma unroll
    for (int r = 0; r < 8; ++r) {
      float mnew = fmaxf(mrun[r], rm[r]);
      alpha[r] = __expf(mrun[r] - mnew);
      mrun[r] = mnew;
      float psum = 0.f;
#pragma unroll
      for (int nt = 0; nt < 2; ++nt) {
        float p = __expf(sc[nt][r] - mnew);
        Ps[wave][half * 8 + r][nt * 16 + nn] = (__bf16)p;
        psum += p;
      }
      rs[r] = psum;
    }
#pragma unroll
    for (int msk = 1; msk <= 8; msk <<= 1)
#pragma unroll
      for (int r = 0; r < 8; ++r)
        rs[r] += __shfl_xor(rs[r], msk, 32);
#pragma unroll
    for (int r = 0; r < 8; ++r) {
      lrun[r] = lrun[r] * alpha[r] + rs[r];
#pragma unroll
      for (int nt = 0; nt < 4; ++nt) o[nt][r] *= alpha[r];
    }
    __syncthreads();

    // O += P @ V  (contraction over 32 keys = one WMMA K-step)
    v16bf pa = load_A_frag(&Ps[wave][0][0], 32 + PAD);
#pragma unroll
    for (int nt = 0; nt < 4; ++nt) {
      v16bf vb = load_B_frag(&Vt[nt * 16][0], 32 + PAD);
      o[nt] = __builtin_amdgcn_wmma_f32_16x16x32_bf16(
          false, pa, false, vb, (short)0, o[nt], false, false);
    }
    __syncthreads();
  }

  // epilogue: normalize and emit bf16 [S][N_HEADS*HEAD_DIM]
#pragma unroll
  for (int r = 0; r < 8; ++r) {
    int qg = qbase + wave * 16 + half * 8 + r;
    float inv = 1.0f / lrun[r];
#pragma unroll
    for (int nt = 0; nt < 4; ++nt)
      Out[(size_t)qg * D_MODEL + h * HEAD_DIM + nt * 16 + nn] =
          (__bf16)(o[nt][r] * inv);
  }
}

// ---------------------------------------------------------------------------
extern "C" void kernel_launch(void* const* d_in, const int* in_sizes, int n_in,
                              void* d_out, int out_size, void* d_ws, size_t ws_size,
                              hipStream_t stream) {
  const float* X  = (const float*)d_in[0];  // [1,2048,2048]
  const float* cf = (const float*)d_in[1];  // [2048,32]
  const float* sf = (const float*)d_in[2];  // [2048,32]
  const float* wq = (const float*)d_in[3];  // [2048,2048]
  const float* wk = (const float*)d_in[4];  // [2048,512]
  const float* wv = (const float*)d_in[5];  // [2048,512]
  const float* wo = (const float*)d_in[6];  // [2048,2048]
  float* out = (float*)d_out;

  // workspace carve-up (256B aligned)
  char* ws = (char*)d_ws;
  size_t off = 0;
  auto carve = [&](size_t bytes) -> __bf16* {
    off = (off + 255) & ~(size_t)255;
    __bf16* p = (__bf16*)(ws + off);
    off += bytes;
    return p;
  };
  const size_t SZ_DD = (size_t)D_MODEL * D_MODEL * 2;  // 8 MB
  const size_t SZ_DK = (size_t)D_MODEL * KV_DIM * 2;   // 2 MB
  __bf16* Xb  = carve(SZ_DD);
  __bf16* Wqb = carve(SZ_DD);
  __bf16* Wkb = carve(SZ_DK);
  __bf16* Wvb = carve(SZ_DK);
  __bf16* Wob = carve(SZ_DD);
  __bf16* Qb  = carve(SZ_DD);
  __bf16* Kb  = carve(SZ_DK);
  __bf16* Vb  = carve(SZ_DK);
  __bf16* Ab  = carve(SZ_DD);

  auto cvt = [&](const float* src, __bf16* dst, int n) {
    cvt_kernel<<<(n + 255) / 256, 256, 0, stream>>>(src, dst, n);
  };
  cvt(X,  Xb,  D_MODEL * D_MODEL);
  cvt(wq, Wqb, D_MODEL * D_MODEL);
  cvt(wk, Wkb, D_MODEL * KV_DIM);
  cvt(wv, Wvb, D_MODEL * KV_DIM);
  cvt(wo, Wob, D_MODEL * D_MODEL);

  // projections: Q = X@Wq [2048x2048], K/V = X@Wk/v [2048x512]
  dim3 gq(D_MODEL / NBLK, SEQ / MBLK);
  dim3 gkv(KV_DIM / NBLK, SEQ / MBLK);
  gemm_bf16<__bf16><<<gq,  256, 0, stream>>>(Xb, Wqb, Qb, SEQ, D_MODEL, D_MODEL);
  gemm_bf16<__bf16><<<gkv, 256, 0, stream>>>(Xb, Wkb, Kb, SEQ, KV_DIM, D_MODEL);
  gemm_bf16<__bf16><<<gkv, 256, 0, stream>>>(Xb, Wvb, Vb, SEQ, KV_DIM, D_MODEL);

  // RoPE on Q and K
  {
    int nq = SEQ * N_HEADS * (HEAD_DIM / 2);
    int nk = SEQ * N_KV * (HEAD_DIM / 2);
    rope_kernel<<<(nq + 255) / 256, 256, 0, stream>>>(Qb, cf, sf, N_HEADS);
    rope_kernel<<<(nk + 255) / 256, 256, 0, stream>>>(Kb, cf, sf, N_KV);
  }

  // attention
  dim3 gfa(SEQ / 128, N_HEADS);
  flash_attn<<<gfa, 256, 0, stream>>>(Qb, Kb, Vb, Ab);

  // output projection (f32 out)
  gemm_bf16<float><<<gq, 256, 0, stream>>>(Ab, Wob, out, SEQ, D_MODEL, D_MODEL);
}
